// End2EndModelSpeaker_50680614093165
// MI455X (gfx1250) — compile-verified, matrix operand order checked by reading, not run on hardware
//
#include <hip/hip_runtime.h>
#include <hip/hip_bf16.h>
#include <cstdint>
#include <cstddef>

// ---------------------------------------------------------------------------
// Model dims
// ---------------------------------------------------------------------------
#define LL 64
#define BB 32
#define WN 50
#define EE 300
#define FF 50
#define DM 100
#define DG 150
#define DP 150
#define DE 100
#define NC 7
#define PP 2

typedef __attribute__((ext_vector_type(16))) _Float16 v16h;
typedef __attribute__((ext_vector_type(8)))  _Float16 v8h;
typedef __attribute__((ext_vector_type(8)))  float    v8f;

// ---------------------------------------------------------------------------
// WMMA helpers (gfx1250: v_wmma_f32_16x16x32_f16, wave32)
// A layout (16-bit, 16x32): lane l holds M = l&15; v16h element j holds
//   K = (l&16 ? 8:0) + (j&8 ? 16:0) + (j&7).   B mirrors with lane->N.
// C/D: lane l holds N = l&15, VGPR v holds M = v + (l&16 ? 8:0).
// ---------------------------------------------------------------------------
__device__ __forceinline__ v8f wmma16(v16h a, v16h b, v8f c) {
  return __builtin_amdgcn_wmma_f32_16x16x32_f16(false, a, false, b, (short)0, c,
                                                false, false);
}

// Build an A fragment from an LDS f16 matrix (row-major, strideH halves,
// strideH multiple of 8, k0 multiple of 32 -> two ds_load_b128 per lane).
__device__ __forceinline__ v16h ldsA(const _Float16* base, int strideH, int m0,
                                     int k0, int lane) {
  const _Float16* p =
      base + (size_t)(m0 + (lane & 15)) * strideH + k0 + ((lane & 16) ? 8 : 0);
  v8h lo = *(const v8h*)(p);
  v8h hi = *(const v8h*)(p + 16);
  v16h a;
#pragma unroll
  for (int j = 0; j < 8; ++j) { a[j] = lo[j]; a[j + 8] = hi[j]; }
  return a;
}

// B fragment from pre-packed weights: packed[(nt*kchunks+kc)*32 + lane][16].
__device__ __forceinline__ v16h pkB(const _Float16* pk, int nt, int kc,
                                    int kchunks, int lane) {
  return *(const v16h*)(pk + (size_t)((nt * kchunks + kc) * 32 + lane) * 16);
}

// Scatter a C tile (f32) to row-major memory (global or LDS).
__device__ __forceinline__ void storeC(float* dst, int strideF, int m0, int n0,
                                       int lane, v8f c) {
  int n = n0 + (lane & 15);
  int mb = m0 + ((lane & 16) ? 8 : 0);
#pragma unroll
  for (int v = 0; v < 8; ++v) dst[(size_t)(mb + v) * strideF + n] = c[v];
}

__device__ __forceinline__ float sigm(float x) { return 1.f / (1.f + expf(-x)); }

// ---------------------------------------------------------------------------
// Weight packer: f32 (N x K, arbitrary strides) -> f16 WMMA-B tile layout,
// zero-padded in both N and K (so A-side padding garbage is harmless).
// ---------------------------------------------------------------------------
__global__ void pack_b_kernel(const float* __restrict__ src,
                              _Float16* __restrict__ dst, int N, int K,
                              int rowStride, int colStride, int ntiles,
                              int kchunks) {
  int idx = blockIdx.x * blockDim.x + threadIdx.x;
  int total = ntiles * kchunks * 512;
  if (idx >= total) return;
  int nt = idx / (kchunks * 512);
  int rem = idx - nt * kchunks * 512;
  int kc = rem / 512;
  int li = rem - kc * 512;
  int lane = li >> 4;
  int j = li & 15;
  int n = nt * 16 + (lane & 15);
  int kk = kc * 32 + ((lane & 16) ? 8 : 0) + ((j & 8) ? 16 : 0) + (j & 7);
  float v = 0.f;
  if (n < N && kk < K) v = src[(size_t)n * rowStride + (size_t)kk * colStride];
  dst[idx] = (_Float16)v;
}

// ---------------------------------------------------------------------------
// Kernel 1: embedding gather -> LDS f16, conv(k=2,3,4) as per-dw WMMA GEMMs,
// relu + max-pool over time, FC 150->100 + relu + utterance mask.
// One workgroup (8 waves) per utterance.
// ---------------------------------------------------------------------------
#define EROWS 68     // 64 padded M positions + max dw(3), rows >=50 zeroed
#define ESTRIDE 320  // E=300 padded to 10 K-chunks

__global__ __launch_bounds__(256) void cnn_feat_kernel(
    const int* __restrict__ x, const float* __restrict__ emb,
    const _Float16* __restrict__ pk_conv, const float* __restrict__ cb2,
    const float* __restrict__ cb3, const float* __restrict__ cb4,
    const float* __restrict__ fc_w, const float* __restrict__ fc_b,
    const float* __restrict__ umask, float* __restrict__ feat) {
  __shared__ _Float16 se[EROWS * ESTRIDE];
  __shared__ float pooled[160];
  __shared__ int words[WN];

  int utt = blockIdx.x;            // utt = l*B + b  (matches reshape(L*B, ...))
  int l = utt >> 5, b = utt & 31;
  int tid = threadIdx.x;

  if (tid < WN) words[tid] = x[utt * WN + tid];
  if (tid < 160) pooled[tid] = 0.f;
  __syncthreads();

  for (int i = tid; i < EROWS * ESTRIDE; i += 256) {
    int r = i / ESTRIDE, c = i - r * ESTRIDE;
    float v = 0.f;
    if (r < WN && c < EE) v = emb[(size_t)words[r] * EE + c];
    se[i] = (_Float16)v;
  }
  __syncthreads();

  int wv = tid >> 5, lane = tid & 31;
  // 12 work items = 3 kernel sizes x 4 N-tiles, round-robin over 8 waves
  for (int wi = wv; wi < 12; wi += 8) {
    int kt = wi >> 2, nt = wi & 3;
    int k = kt + 2;
    int slice0 = (kt == 0) ? 0 : (kt == 1) ? 2 : 5;  // dw-slices base
    const float* bias = (kt == 0) ? cb2 : (kt == 1) ? cb3 : cb4;
    int tvalid = WN - k + 1;
    int n = nt * 16 + (lane & 15);
    float bv = (n < FF) ? bias[n] : 0.f;
    for (int mt = 0; mt < 4; ++mt) {
      v8f acc = {};
      for (int dw = 0; dw < k; ++dw) {
        const _Float16* pk = pk_conv + (size_t)(slice0 + dw) * (4 * 10 * 512);
        for (int kc = 0; kc < 10; ++kc) {
          v16h a = ldsA(se, ESTRIDE, mt * 16 + dw, kc * 32, lane);
          acc = wmma16(a, pkB(pk, nt, kc, 10, lane), acc);
        }
      }
      int mb = mt * 16 + ((lane & 16) ? 8 : 0);
      float mx = -1.f;
#pragma unroll
      for (int v = 0; v < 8; ++v) {
        float val = acc[v] + bv;
        val = val > 0.f ? val : 0.f;
        if (mb + v < tvalid) mx = mx > val ? mx : val;
      }
      if (n < FF && mx >= 0.f)  // nonneg floats order-preserve as ints
        atomicMax((int*)&pooled[kt * FF + n], __float_as_int(mx));
    }
  }
  __syncthreads();

  if (tid < DM) {  // FC 150 -> 100 (tiny; VALU)
    float s = fc_b[tid];
    for (int i = 0; i < 3 * FF; ++i) s += fc_w[tid * (3 * FF) + i] * pooled[i];
    s = s > 0.f ? s : 0.f;
    s *= umask[b * LL + l];
    feat[(size_t)utt * DM + tid] = s;
  }
}

// ---------------------------------------------------------------------------
// Kernel 2: DialogueRNN scan. grid.x = direction (0 fwd / 1 bwd flipped).
// One persistent 512-thread (16-wave) workgroup per direction; all state in
// dynamic LDS; every x@W^T is WMMA; gates/attention are VALU.
// ---------------------------------------------------------------------------
struct DirParams {
  const _Float16 *g_wih, *g_whh, *p_wih, *p_whh, *e_wih, *e_whh, *att_w;
  const float *g_bih, *g_bhh, *p_bih, *p_bhh, *e_bih, *e_bhh, *att_b;
  float *g_hist, *e_seq, *gi_s, *gh_s;  // workspace (per-direction)
};

#define SCAN_SMEM_BYTES                                                        \
  ((64 * 256 + 64 * 160 + 32 * 160 + 32 * 128 + 32 * 160) * 2 +               \
   (64 * 150 + 32 * 150 + 32 * 100 + 32 * 100 + 32 * 160 + 32 * 64 +          \
    32 * 150) * 4 +                                                            \
   32 * 4)

__global__ __launch_bounds__(512) void drnn_scan_kernel(
    const float* __restrict__ feat, const float* __restrict__ qmask,
    DirParams PF, DirParams PR) {
  extern __shared__ char smem[];
  _Float16* xh = (_Float16*)smem;     // 64x256: x rows(32) / uc_p rows(64), f16
  _Float16* qh = xh + 64 * 256;       // 64x160: party state q, f16 (K-padded)
  _Float16* gph = qh + 64 * 160;      // 32x160: g_prev, f16
  _Float16* eh = gph + 32 * 160;      // 32x128: e_prev, f16
  _Float16* esh = eh + 32 * 128;      // 32x160: qn[b,qidx], f16
  float* qf = (float*)(esh + 32 * 160);  // 64x150 q f32
  float* gf = qf + 64 * 150;             // 32x150 g f32
  float* ef = gf + 32 * 150;             // 32x100 e f32
  float* uf = ef + 32 * 100;             // 32x100 u f32
  float* xt = uf + 32 * 100;             // 32x160 attention query
  float* sc = xt + 32 * 160;             // 32x64  scores/alpha
  float* cb = sc + 32 * 64;              // 32x150 context
  int* qidx = (int*)(cb + 32 * 150);     // 32

  const int dir = blockIdx.x;
  const DirParams P = dir ? PR : PF;
  const int tid = threadIdx.x;
  const int wv = tid >> 5, lane = tid & 31;

  for (int i = tid; i < SCAN_SMEM_BYTES / 4; i += 512) ((int*)smem)[i] = 0;
  __syncthreads();

  for (int t = 0; t < LL; ++t) {
    const int tt = dir ? (LL - 1 - t) : t;

    // phase 1: load u, speaker index, build x = [u, q_sel] (f16)
    for (int i = tid; i < BB * DM; i += 512)
      uf[i] = feat[(size_t)(tt * BB) * DM + i];
    if (tid < BB) {
      const float* qm = qmask + (size_t)(tt * BB + tid) * PP;
      qidx[tid] = (qm[1] > qm[0]) ? 1 : 0;
    }
    __syncthreads();
    for (int i = tid; i < BB * 256; i += 512) {
      int b = i >> 8, d = i & 255;
      float v = 0.f;
      if (d < DM) v = uf[b * DM + d];
      else if (d < DM + DP) v = qf[(b * PP + qidx[b]) * DP + (d - DM)];
      xh[i] = (_Float16)v;
    }
    __syncthreads();

    // phase 2: gi = x @ g_wih^T, gh = g_prev @ g_whh^T   (WMMA)
    for (int wi = wv; wi < 2 * 2 * 29; wi += 16) {
      int which = wi >= 58;
      int r = which ? wi - 58 : wi;
      int mt = r / 29, nt = r - mt * 29;
      v8f acc = {};
      if (!which) {
        for (int kc = 0; kc < 8; ++kc)
          acc = wmma16(ldsA(xh, 256, mt * 16, kc * 32, lane),
                       pkB(P.g_wih, nt, kc, 8, lane), acc);
        storeC(P.gi_s, 464, mt * 16, nt * 16, lane, acc);
      } else {
        for (int kc = 0; kc < 5; ++kc)
          acc = wmma16(ldsA(gph, 160, mt * 16, kc * 32, lane),
                       pkB(P.g_whh, nt, kc, 5, lane), acc);
        storeC(P.gh_s, 464, mt * 16, nt * 16, lane, acc);
      }
    }
    __syncthreads();

    // phase 3: g GRU gate combine; publish g_hist[t]
    for (int i = tid; i < BB * DG; i += 512) {
      int b = i / DG, d = i - b * DG;
      const float* gi = P.gi_s + (size_t)b * 464;
      const float* gh = P.gh_s + (size_t)b * 464;
      float rr = sigm(gi[d] + P.g_bih[d] + gh[d] + P.g_bhh[d]);
      float zz = sigm(gi[DG + d] + P.g_bih[DG + d] + gh[DG + d] + P.g_bhh[DG + d]);
      float hn = gh[2 * DG + d] + P.g_bhh[2 * DG + d];
      float nn = tanhf(gi[2 * DG + d] + P.g_bih[2 * DG + d] + rr * hn);
      float h = gf[i];
      float gn = (1.f - zz) * nn + zz * h;
      gf[i] = gn;
      gph[b * 160 + d] = (_Float16)gn;
      P.g_hist[(size_t)(t * BB + b) * DG + d] = gn;
    }
    __syncthreads();

    // phase 4: attention xt = u @ att_w^T (WMMA; K pad cols of B are zero)
    for (int wi = wv; wi < 2 * 10; wi += 16) {
      int mt = wi / 10, nt = wi - mt * 10;
      v8f acc = {};
      for (int kc = 0; kc < 4; ++kc)
        acc = wmma16(ldsA(xh, 256, mt * 16, kc * 32, lane),
                     pkB(P.att_w, nt, kc, 4, lane), acc);
      storeC(xt, 160, mt * 16, nt * 16, lane, acc);
    }
    __syncthreads();
    for (int i = tid; i < BB * DG; i += 512) {
      int b = i / DG, d = i - b * DG;
      xt[b * 160 + d] += P.att_b[d];
    }
    __syncthreads();
    for (int i = tid; i < BB * t; i += 512) {  // scores over history l < t
      int b = i & 31, hl = i >> 5;
      const float* gh = P.g_hist + (size_t)(hl * BB + b) * DG;
      const float* xb = xt + b * 160;
      float s = 0.f;
      for (int d = 0; d < DG; ++d) s += xb[d] * gh[d];
      sc[b * 64 + hl] = s;
    }
    __syncthreads();
    if (tid < BB && t > 0) {  // masked softmax per dialogue
      float mx = -1e30f;
      for (int hl = 0; hl < t; ++hl) mx = fmaxf(mx, sc[tid * 64 + hl]);
      float sum = 0.f;
      for (int hl = 0; hl < t; ++hl) {
        float e = expf(sc[tid * 64 + hl] - mx);
        sc[tid * 64 + hl] = e;
        sum += e;
      }
      float inv = 1.f / sum;
      for (int hl = 0; hl < t; ++hl) sc[tid * 64 + hl] *= inv;
    }
    __syncthreads();
    for (int i = tid; i < BB * DG; i += 512) {  // context (zero at t==0)
      int b = i / DG, d = i - b * DG;
      float s = 0.f;
      for (int hl = 0; hl < t; ++hl)
        s += sc[b * 64 + hl] * P.g_hist[(size_t)(hl * BB + b) * DG + d];
      cb[i] = s;
    }
    __syncthreads();

    // phase 5: uc_p rows (b,p) = [u(b), c(b)]; p GRU GEMMs
    for (int i = tid; i < 64 * 256; i += 512) {
      int rp = i >> 8, d = i & 255;
      int b = rp >> 1;
      float v = 0.f;
      if (d < DM) v = uf[b * DM + d];
      else if (d < DM + DG) v = cb[b * DG + (d - DM)];
      xh[i] = (_Float16)v;
    }
    __syncthreads();
    for (int wi = wv; wi < 2 * 4 * 29; wi += 16) {
      int which = wi >= 116;
      int r = which ? wi - 116 : wi;
      int mt = r / 29, nt = r - mt * 29;
      v8f acc = {};
      if (!which) {
        for (int kc = 0; kc < 8; ++kc)
          acc = wmma16(ldsA(xh, 256, mt * 16, kc * 32, lane),
                       pkB(P.p_wih, nt, kc, 8, lane), acc);
        storeC(P.gi_s, 464, mt * 16, nt * 16, lane, acc);
      } else {
        for (int kc = 0; kc < 5; ++kc)
          acc = wmma16(ldsA(qh, 160, mt * 16, kc * 32, lane),
                       pkB(P.p_whh, nt, kc, 5, lane), acc);
        storeC(P.gh_s, 464, mt * 16, nt * 16, lane, acc);
      }
    }
    __syncthreads();

    // phase 6: p GRU combine; update only speaker rows (listener_state=False)
    for (int i = tid; i < 64 * DP; i += 512) {
      int rp = i / DP, d = i - rp * DP;
      int b = rp >> 1, p = rp & 1;
      const float* gi = P.gi_s + (size_t)rp * 464;
      const float* gh = P.gh_s + (size_t)rp * 464;
      float rr = sigm(gi[d] + P.p_bih[d] + gh[d] + P.p_bhh[d]);
      float zz = sigm(gi[DP + d] + P.p_bih[DP + d] + gh[DP + d] + P.p_bhh[DP + d]);
      float hn = gh[2 * DP + d] + P.p_bhh[2 * DP + d];
      float nn = tanhf(gi[2 * DP + d] + P.p_bih[2 * DP + d] + rr * hn);
      float h = qf[i];
      float qs = (1.f - zz) * nn + zz * h;
      if (p == qidx[b]) {
        qf[i] = qs;
        qh[rp * 160 + d] = (_Float16)qs;
        esh[b * 160 + d] = (_Float16)qs;  // e-GRU input = qn[b, qidx]
      }
    }
    __syncthreads();

    // phase 7: e GRU GEMMs
    for (int wi = wv; wi < 2 * 2 * 19; wi += 16) {
      int which = wi >= 38;
      int r = which ? wi - 38 : wi;
      int mt = r / 19, nt = r - mt * 19;
      v8f acc = {};
      if (!which) {
        for (int kc = 0; kc < 5; ++kc)
          acc = wmma16(ldsA(esh, 160, mt * 16, kc * 32, lane),
                       pkB(P.e_wih, nt, kc, 5, lane), acc);
        storeC(P.gi_s, 464, mt * 16, nt * 16, lane, acc);
      } else {
        for (int kc = 0; kc < 4; ++kc)
          acc = wmma16(ldsA(eh, 128, mt * 16, kc * 32, lane),
                       pkB(P.e_whh, nt, kc, 4, lane), acc);
        storeC(P.gh_s, 464, mt * 16, nt * 16, lane, acc);
      }
    }
    __syncthreads();

    // phase 8: e GRU combine; emit e_seq at original turn index tt
    for (int i = tid; i < BB * DE; i += 512) {
      int b = i / DE, d = i - b * DE;
      const float* gi = P.gi_s + (size_t)b * 464;
      const float* gh = P.gh_s + (size_t)b * 464;
      float rr = sigm(gi[d] + P.e_bih[d] + gh[d] + P.e_bhh[d]);
      float zz = sigm(gi[DE + d] + P.e_bih[DE + d] + gh[DE + d] + P.e_bhh[DE + d]);
      float hn = gh[2 * DE + d] + P.e_bhh[2 * DE + d];
      float nn = tanhf(gi[2 * DE + d] + P.e_bih[2 * DE + d] + rr * hn);
      float h = ef[i];
      float en = (1.f - zz) * nn + zz * h;
      ef[i] = en;
      eh[b * 128 + d] = (_Float16)en;
      P.e_seq[(size_t)(tt * BB + b) * DE + d] = en;
    }
    __syncthreads();
  }
}

// ---------------------------------------------------------------------------
// Kernel 3: hidden = relu([e_f,e_b] @ lin_w^T + lin_b) via WMMA,
// then 7-way logits + log_softmax (VALU). One wave per 16-row M-tile.
// ---------------------------------------------------------------------------
__global__ __launch_bounds__(32) void classifier_kernel(
    const float* __restrict__ e_f, const float* __restrict__ e_b,
    const _Float16* __restrict__ pk_lin, const float* __restrict__ lin_b,
    const float* __restrict__ smax_w, const float* __restrict__ smax_b,
    float* __restrict__ out) {
  __shared__ float hid[16][112];
  int mt = blockIdx.x;
  int lane = threadIdx.x;
  int m = mt * 16 + (lane & 15);

  v8f acc[7];
#pragma unroll
  for (int nt = 0; nt < 7; ++nt) acc[nt] = (v8f){};

  for (int kc = 0; kc < 7; ++kc) {
    v16h a;
#pragma unroll
    for (int j = 0; j < 16; ++j) {
      int k = kc * 32 + ((lane & 16) ? 8 : 0) + ((j & 8) ? 16 : 0) + (j & 7);
      float v = 0.f;
      if (k < DE) v = e_f[(size_t)m * DE + k];
      else if (k < 2 * DE) v = e_b[(size_t)m * DE + (k - DE)];
      a[j] = (_Float16)v;
    }
#pragma unroll
    for (int nt = 0; nt < 7; ++nt)
      acc[nt] = wmma16(a, pkB(pk_lin, nt, kc, 7, lane), acc[nt]);
  }
#pragma unroll
  for (int nt = 0; nt < 7; ++nt) {
    int n = nt * 16 + (lane & 15);
    int mb = (lane & 16) ? 8 : 0;
    float bv = (n < DM) ? lin_b[n] : 0.f;
#pragma unroll
    for (int v = 0; v < 8; ++v) {
      float val = acc[nt][v] + bv;
      hid[mb + v][n] = val > 0.f ? val : 0.f;
    }
  }
  __syncthreads();

  if (lane < 16) {
    int row = mt * 16 + lane;
    float logit[NC];
    float mx = -1e30f;
    for (int c = 0; c < NC; ++c) {
      float s = smax_b[c];
      for (int d = 0; d < DM; ++d) s += hid[lane][d] * smax_w[c * DM + d];
      logit[c] = s;
      mx = fmaxf(mx, s);
    }
    float sum = 0.f;
    for (int c = 0; c < NC; ++c) sum += expf(logit[c] - mx);
    float lse = mx + logf(sum);
    for (int c = 0; c < NC; ++c) out[(size_t)row * NC + c] = logit[c] - lse;
  }
}

// ---------------------------------------------------------------------------
// Host launcher. Input order = JAX pytree flatten (dict keys sorted).
// ---------------------------------------------------------------------------
enum {
  IW2 = 0, IB2, IW3, IB3, IW4, IB4, IEMB,
  IF_ATT_B, IF_ATT_W, IF_E_BHH, IF_E_BIH, IF_E_WHH, IF_E_WIH,
  IF_G_BHH, IF_G_BIH, IF_G_WHH, IF_G_WIH,
  IF_P_BHH, IF_P_BIH, IF_P_WHH, IF_P_WIH,
  IFC_B, IFC_W, ILIN_B, ILIN_W,
  IR_ATT_B, IR_ATT_W, IR_E_BHH, IR_E_BIH, IR_E_WHH, IR_E_WIH,
  IR_G_BHH, IR_G_BIH, IR_G_WHH, IR_G_WIH,
  IR_P_BHH, IR_P_BIH, IR_P_WHH, IR_P_WIH,
  ISMAX_B, ISMAX_W, IX, IQMASK, IUMASK
};

extern "C" void kernel_launch(void* const* d_in, const int* in_sizes, int n_in,
                              void* d_out, int out_size, void* d_ws,
                              size_t ws_size, hipStream_t stream) {
  (void)in_sizes; (void)n_in; (void)out_size; (void)ws_size;

  char* w = (char*)d_ws;
  size_t off = 0;
  auto take = [&](size_t bytes) -> char* {
    char* p = w + off;
    off = (off + bytes + 255) & ~(size_t)255;
    return p;
  };
  auto pack = [&](const float* src, char* dst, int N, int K, int rowStride,
                  int colStride) {
    int ntiles = (N + 15) / 16, kch = (K + 31) / 32;
    int total = ntiles * kch * 512;
    pack_b_kernel<<<dim3((total + 255) / 256), dim3(256), 0, stream>>>(
        src, (_Float16*)dst, N, K, rowStride, colStride, ntiles, kch);
  };

  // ---- workspace layout -------------------------------------------------
  char* pk_conv = take((size_t)9 * 4 * 10 * 512 * 2);  // 9 dw-slices
  char* pk_gwih[2], *pk_gwhh[2], *pk_pwih[2], *pk_pwhh[2];
  char* pk_ewih[2], *pk_ewhh[2], *pk_att[2];
  for (int d = 0; d < 2; ++d) {
    pk_gwih[d] = take((size_t)29 * 8 * 512 * 2);
    pk_gwhh[d] = take((size_t)29 * 5 * 512 * 2);
    pk_pwih[d] = take((size_t)29 * 8 * 512 * 2);
    pk_pwhh[d] = take((size_t)29 * 5 * 512 * 2);
    pk_ewih[d] = take((size_t)19 * 5 * 512 * 2);
    pk_ewhh[d] = take((size_t)19 * 4 * 512 * 2);
    pk_att[d]  = take((size_t)10 * 4 * 512 * 2);
  }
  char* pk_lin = take((size_t)7 * 7 * 512 * 2);
  char* feat   = take((size_t)LL * BB * DM * 4);
  char* g_hist[2], *e_seq[2], *gi_s[2], *gh_s[2];
  for (int d = 0; d < 2; ++d) {
    g_hist[d] = take((size_t)LL * BB * DG * 4);
    e_seq[d]  = take((size_t)LL * BB * DE * 4);
    gi_s[d]   = take((size_t)64 * 464 * 4);
    gh_s[d]   = take((size_t)64 * 464 * 4);
  }

  // ---- pack all weights to f16 WMMA-B layout ----------------------------
  const float* cw[3] = {(const float*)d_in[IW2], (const float*)d_in[IW3],
                        (const float*)d_in[IW4]};
  int slice = 0;
  for (int kt = 0; kt < 3; ++kt) {
    int k = kt + 2;
    for (int dw = 0; dw < k; ++dw, ++slice)
      pack(cw[kt] + dw, pk_conv + (size_t)slice * 4 * 10 * 512 * 2, FF, EE,
           EE * k, k);  // w[f, e, dw] slice: rowStride=E*k, colStride=k
  }
  int base[2] = {IF_ATT_B, IR_ATT_B};
  for (int d = 0; d < 2; ++d) {
    int o = base[d];
    pack((const float*)d_in[o + 9], pk_gwih[d], 3 * DG, DM + DP, DM + DP, 1);
    pack((const float*)d_in[o + 8], pk_gwhh[d], 3 * DG, DG, DG, 1);
    pack((const float*)d_in[o + 13], pk_pwih[d], 3 * DP, DM + DG, DM + DG, 1);
    pack((const float*)d_in[o + 12], pk_pwhh[d], 3 * DP, DP, DP, 1);
    pack((const float*)d_in[o + 5], pk_ewih[d], 3 * DE, DP, DP, 1);
    pack((const float*)d_in[o + 4], pk_ewhh[d], 3 * DE, DE, DE, 1);
    pack((const float*)d_in[o + 1], pk_att[d], DG, DM, DM, 1);
  }
  pack((const float*)d_in[ILIN_W], pk_lin, DM, 2 * DE, 2 * DE, 1);

  // ---- CNN features -----------------------------------------------------
  cnn_feat_kernel<<<dim3(LL * BB), dim3(256), 0, stream>>>(
      (const int*)d_in[IX], (const float*)d_in[IEMB], (const _Float16*)pk_conv,
      (const float*)d_in[IB2], (const float*)d_in[IB3],
      (const float*)d_in[IB4], (const float*)d_in[IFC_W],
      (const float*)d_in[IFC_B], (const float*)d_in[IUMASK], (float*)feat);

  // ---- bidirectional DialogueRNN scan -----------------------------------
  DirParams DPm[2];
  char* pkw[2][7] = {{pk_gwih[0], pk_gwhh[0], pk_pwih[0], pk_pwhh[0],
                      pk_ewih[0], pk_ewhh[0], pk_att[0]},
                     {pk_gwih[1], pk_gwhh[1], pk_pwih[1], pk_pwhh[1],
                      pk_ewih[1], pk_ewhh[1], pk_att[1]}};
  for (int d = 0; d < 2; ++d) {
    int o = base[d];
    DPm[d].g_wih = (const _Float16*)pkw[d][0];
    DPm[d].g_whh = (const _Float16*)pkw[d][1];
    DPm[d].p_wih = (const _Float16*)pkw[d][2];
    DPm[d].p_whh = (const _Float16*)pkw[d][3];
    DPm[d].e_wih = (const _Float16*)pkw[d][4];
    DPm[d].e_whh = (const _Float16*)pkw[d][5];
    DPm[d].att_w = (const _Float16*)pkw[d][6];
    DPm[d].g_bih = (const float*)d_in[o + 7];
    DPm[d].g_bhh = (const float*)d_in[o + 6];
    DPm[d].p_bih = (const float*)d_in[o + 11];
    DPm[d].p_bhh = (const float*)d_in[o + 10];
    DPm[d].e_bih = (const float*)d_in[o + 3];
    DPm[d].e_bhh = (const float*)d_in[o + 2];
    DPm[d].att_b = (const float*)d_in[o + 0];
    DPm[d].g_hist = (float*)g_hist[d];
    DPm[d].e_seq = (float*)e_seq[d];
    DPm[d].gi_s = (float*)gi_s[d];
    DPm[d].gh_s = (float*)gh_s[d];
  }
  drnn_scan_kernel<<<dim3(2), dim3(512), SCAN_SMEM_BYTES, stream>>>(
      (const float*)feat, (const float*)d_in[IQMASK], DPm[0], DPm[1]);

  // ---- classifier + log_softmax ----------------------------------------
  classifier_kernel<<<dim3(LL * BB / 16), dim3(32), 0, stream>>>(
      (const float*)e_seq[0], (const float*)e_seq[1], (const _Float16*)pk_lin,
      (const float*)d_in[ILIN_B], (const float*)d_in[ISMAX_W],
      (const float*)d_in[ISMAX_B], (float*)d_out);
}
